// MultiHeadedAttention_4329327034668
// MI455X (gfx1250) — matrix operand order repaired and checked by reading, not compiled, not run on hardware
//
#include <hip/hip_runtime.h>
#include <hip/hip_bf16.h>
#include <stdint.h>

typedef __attribute__((ext_vector_type(16))) _Float16 v16h;
typedef __attribute__((ext_vector_type(8)))  _Float16 v8h;
typedef __attribute__((ext_vector_type(8)))  float    v8f;

union V16 { v16h v; v8h h[2]; };

__device__ __forceinline__ v8f vzero8f() {
  v8f z;
#pragma unroll
  for (int i = 0; i < 8; ++i) z[i] = 0.0f;
  return z;
}

__device__ __forceinline__ v8f wmma_f16(const V16& a, const V16& b, v8f c) {
  return __builtin_amdgcn_wmma_f32_16x16x32_f16(
      /*neg_a=*/false, a.v, /*neg_b=*/false, b.v,
      /*c_mod=*/(short)0, c, /*reuse_a=*/false, /*reuse_b=*/false);
}

// Problem dims (compile-time)
constexpr int Bsz  = 2;
constexpr int Lsz  = 2048;
constexpr int Dsz  = 1024;
constexpr int Hsz  = 16;
constexpr int DHsz = 64;
constexpr int Mrows = Bsz * Lsz;  // 4096

// ---------------------------------------------------------------------------
// Tiled WMMA GEMM:  Y = (A @ W + bias) * scale
// Tile 128x128, K-step 32, 256 threads = 8 waves, each wave: 2x4 16x16 tiles.
// ---------------------------------------------------------------------------
template<bool A_F16, bool OUT_F32, bool OUT_T>
__global__ __launch_bounds__(256) void gemm_bias_kernel(
    const void* __restrict__ Aptr, const float* __restrict__ W,
    const float* __restrict__ bias, void* __restrict__ Yptr,
    int M, int N, int K, float scale)
{
  __shared__ __align__(16) _Float16 As[128][40];  // 128 rows x 32 K (+pad)
  __shared__ __align__(16) _Float16 Bs[128][40];  // 128 N (transposed) x 32 K

  const int tid  = threadIdx.x;
  const int lane = tid & 31;
  const int wid  = tid >> 5;
  const int g    = lane >> 4;
  const int lr   = lane & 15;
  const int m0   = blockIdx.y * 128;
  const int n0   = blockIdx.x * 128;
  const int wr   = wid >> 1;
  const int wc   = wid & 1;

  v8f acc[2][4];
#pragma unroll
  for (int mi = 0; mi < 2; ++mi)
#pragma unroll
    for (int ni = 0; ni < 4; ++ni) acc[mi][ni] = vzero8f();

  for (int k0 = 0; k0 < K; k0 += 32) {
    __syncthreads();
#pragma unroll
    for (int i = 0; i < 16; ++i) {
      int idx = tid + i * 256;            // 0..4095
      int r = idx >> 5, c = idx & 31;     // A: row, k
      float a;
      if (A_F16) a = (float)((const _Float16*)Aptr)[(size_t)(m0 + r) * K + k0 + c];
      else       a = ((const float*)Aptr)[(size_t)(m0 + r) * K + k0 + c];
      As[r][c] = (_Float16)a;
      int kk = idx >> 7, n = idx & 127;   // W: k, n (coalesced over n)
      Bs[n][kk] = (_Float16)W[(size_t)(k0 + kk) * N + n0 + n];
    }
    __syncthreads();

    V16 af[2], bf[4];
#pragma unroll
    for (int mi = 0; mi < 2; ++mi) {
      int r = wr * 32 + mi * 16 + lr;
      af[mi].h[0] = *(const v8h*)&As[r][g * 8];
      af[mi].h[1] = *(const v8h*)&As[r][16 + g * 8];
    }
#pragma unroll
    for (int ni = 0; ni < 4; ++ni) {
      int n = wc * 64 + ni * 16 + lr;
      bf[ni].h[0] = *(const v8h*)&Bs[n][g * 8];
      bf[ni].h[1] = *(const v8h*)&Bs[n][16 + g * 8];
    }
#pragma unroll
    for (int mi = 0; mi < 2; ++mi)
#pragma unroll
      for (int ni = 0; ni < 4; ++ni)
        acc[mi][ni] = wmma_f16(af[mi], bf[ni], acc[mi][ni]);
  }

#pragma unroll
  for (int mi = 0; mi < 2; ++mi) {
#pragma unroll
    for (int ni = 0; ni < 4; ++ni) {
      int col = n0 + wc * 64 + ni * 16 + lr;
      float bv = bias[col];
      if (OUT_T) {
        int mbase = m0 + wr * 32 + mi * 16 + g * 8;
        v8h pk;
#pragma unroll
        for (int j = 0; j < 8; ++j) pk[j] = (_Float16)((acc[mi][ni][j] + bv) * scale);
        *(v8h*)((_Float16*)Yptr + (size_t)col * M + mbase) = pk;
      } else {
#pragma unroll
        for (int j = 0; j < 8; ++j) {
          int m = m0 + wr * 32 + mi * 16 + j + g * 8;
          float y = (acc[mi][ni][j] + bv) * scale;
          if (OUT_F32) ((float*)Yptr)[(size_t)m * N + col] = y;
          else         ((_Float16*)Yptr)[(size_t)m * N + col] = (_Float16)y;
        }
      }
    }
  }
}

// ---------------------------------------------------------------------------
// Flash attention with TRANSPOSED score tiles: S^T = K . Q^T, so C rows are
// keys and C cols are queries. Each lane then owns one query column and 8
// consecutive keys per VGPR:
//   - softmax over keys: in-lane reduction + one shfl_xor(16)
//   - mask: one uint64 (8-byte) load per 16-key tile
//   - head-0 raw scores: one v8f store per tile
//   - P^T restage to LDS: packed v8h stores
// ctx is accumulated transposed (ctx^T = V^T . P^T), giving contiguous A-frag
// loads from the pre-transposed V buffer and packed v8h ctx stores.
// 128 threads = 4 waves, each wave owns 16 query rows, streams 32-key tiles.
// ---------------------------------------------------------------------------
__global__ __launch_bounds__(128) void flash_attn_kernel(
    const _Float16* __restrict__ Q, const _Float16* __restrict__ Kmat,
    const _Float16* __restrict__ Vt, const unsigned char* __restrict__ mask,
    _Float16* __restrict__ ctx, float* __restrict__ attn_raw)
{
  __shared__ __align__(16) _Float16 Pst[4][16][32];  // per-wave P^T: [q][key]

  const int tid  = threadIdx.x;
  const int wid  = tid >> 5;
  const int lane = tid & 31;
  const int g    = lane >> 4;
  const int lr   = lane & 15;
  const int b    = blockIdx.z;
  const int h    = blockIdx.y;
  const int q0   = blockIdx.x * 64 + wid * 16;
  const int qrow = q0 + lr;                          // this lane's query

  const _Float16* Qh = Q    + (size_t)b * Lsz * Dsz + (size_t)h * DHsz;
  const _Float16* Kh = Kmat + (size_t)b * Lsz * Dsz + (size_t)h * DHsz;
  const unsigned char* mrow = mask + ((size_t)b * Lsz + qrow) * Lsz;
  float* arow = attn_raw + ((size_t)b * Lsz + qrow) * Lsz;

  // Preload Q as B-fragments (dh x 16 queries): lane col = query lr,
  // elements = dh 8g..8g+7 and 16+8g..16+8g+7 per half-wave.
  V16 qb[2];
#pragma unroll
  for (int kb = 0; kb < 2; ++kb) {
    qb[kb].h[0] = *(const v8h*)(Qh + (size_t)qrow * Dsz + kb * 32 + g * 8);
    qb[kb].h[1] = *(const v8h*)(Qh + (size_t)qrow * Dsz + kb * 32 + 16 + g * 8);
  }

  float mrun = -3.0e38f, lrun = 0.0f;
  v8f o[4];                                          // ctx^T accum: dh rows, q cols
#pragma unroll
  for (int ni = 0; ni < 4; ++ni) o[ni] = vzero8f();

  for (int kt = 0; kt < Lsz; kt += 32) {
    if (kt + 32 < Lsz)
      __builtin_prefetch((const void*)(Kh + (size_t)(kt + 32 + lr) * Dsz), 0, 1);

    // ---- S^T tiles: A = K rows (16 keys x 64 dh), B = Q^T ----
    v8f S[2];
#pragma unroll
    for (int t = 0; t < 2; ++t) {
      const int key = kt + t * 16 + lr;              // A-frag row = key
      V16 ka0, ka1;
      ka0.h[0] = *(const v8h*)(Kh + (size_t)key * Dsz + g * 8);
      ka0.h[1] = *(const v8h*)(Kh + (size_t)key * Dsz + 16 + g * 8);
      ka1.h[0] = *(const v8h*)(Kh + (size_t)key * Dsz + 32 + g * 8);
      ka1.h[1] = *(const v8h*)(Kh + (size_t)key * Dsz + 32 + 16 + g * 8);
      S[t] = vzero8f();
      S[t] = wmma_f16(ka0, qb[0], S[t]);
      S[t] = wmma_f16(ka1, qb[1], S[t]);
    }

    // ---- mask (8 consecutive key bytes per lane) + head-0 raw scores ----
#pragma unroll
    for (int t = 0; t < 2; ++t) {
      const int kbase = kt + t * 16 + 8 * g;         // this lane's 8 keys
      const uint64_t mb = *(const uint64_t*)(mrow + kbase);
#pragma unroll
      for (int j = 0; j < 8; ++j)
        if ((mb >> (8 * j)) & 0xff) S[t][j] = -1.0e18f;
      if (h == 0) *(v8f*)(arow + kbase) = S[t];      // contiguous 32B store
    }

    // ---- online softmax over keys (per query column) ----
    float mx = S[0][0];
#pragma unroll
    for (int j = 1; j < 8; ++j) mx = fmaxf(mx, S[0][j]);
#pragma unroll
    for (int j = 0; j < 8; ++j) mx = fmaxf(mx, S[1][j]);
    mx = fmaxf(mx, __shfl_xor(mx, 16, 32));          // combine key halves
    const float mnew  = fmaxf(mrun, mx);
    const float alpha = __expf(mrun - mnew);
    mrun = mnew;

    float rs = 0.0f;
    v8h ph[2];
#pragma unroll
    for (int t = 0; t < 2; ++t)
#pragma unroll
      for (int j = 0; j < 8; ++j) {
        const float p = __expf(S[t][j] - mnew);
        rs += p;
        ph[t][j] = (_Float16)p;
      }
    rs += __shfl_xor(rs, 16, 32);
    lrun = lrun * alpha + rs;
#pragma unroll
    for (int ni = 0; ni < 4; ++ni)
#pragma unroll
      for (int j = 0; j < 8; ++j) o[ni][j] *= alpha;

    // ---- restage P^T (q x 32 keys) via LDS, packed v8h stores ----
    *(v8h*)&Pst[wid][lr][8 * g]      = ph[0];
    *(v8h*)&Pst[wid][lr][16 + 8 * g] = ph[1];
    asm volatile("s_wait_dscnt 0x0" ::: "memory");   // intra-wave LDS ordering

    V16 pb;                                          // B-frag: col=q, keys 8g.., 16+8g..
    pb.h[0] = *(const v8h*)&Pst[wid][lr][g * 8];
    pb.h[1] = *(const v8h*)&Pst[wid][lr][16 + g * 8];

    // ---- ctx^T += V^T . P^T  (A-frags contiguous from transposed V) ----
#pragma unroll
    for (int ni = 0; ni < 4; ++ni) {
      const _Float16* vrow =
          Vt + (size_t)(h * DHsz + ni * 16 + lr) * Mrows + (size_t)b * Lsz + kt;
      V16 va;
      va.h[0] = *(const v8h*)(vrow + g * 8);
      va.h[1] = *(const v8h*)(vrow + 16 + g * 8);
      o[ni] = wmma_f16(va, pb, o[ni]);
    }
  }

  // ---- normalize, pack, and store ctx (8 contiguous dh per lane) ----
  const float inv = 1.0f / lrun;
  _Float16* crow = ctx + ((size_t)b * Lsz + qrow) * Dsz + h * DHsz;
#pragma unroll
  for (int ni = 0; ni < 4; ++ni) {
    v8h pk;
#pragma unroll
    for (int j = 0; j < 8; ++j) pk[j] = (_Float16)(o[ni][j] * inv);
    *(v8h*)(crow + ni * 16 + 8 * g) = pk;            // 16B store
  }
}

// ---------------------------------------------------------------------------
// In-place row softmax over head-0 scores (B*L rows of length L) for top_attn.
// ---------------------------------------------------------------------------
__global__ __launch_bounds__(256) void softmax_rows_kernel(float* __restrict__ attn)
{
  __shared__ float sred[8];
  float* p = attn + (size_t)blockIdx.x * Lsz;
  const int tid = threadIdx.x;

  float mx = -3.0e38f;
  for (int i = tid; i < Lsz; i += 256) mx = fmaxf(mx, p[i]);
#pragma unroll
  for (int off = 16; off > 0; off >>= 1) mx = fmaxf(mx, __shfl_xor(mx, off, 32));
  if ((tid & 31) == 0) sred[tid >> 5] = mx;
  __syncthreads();
  float bm = sred[0];
#pragma unroll
  for (int w = 1; w < 8; ++w) bm = fmaxf(bm, sred[w]);
  __syncthreads();

  float sum = 0.0f;
  for (int i = tid; i < Lsz; i += 256) sum += __expf(p[i] - bm);
#pragma unroll
  for (int off = 16; off > 0; off >>= 1) sum += __shfl_xor(sum, off, 32);
  if ((tid & 31) == 0) sred[tid >> 5] = sum;
  __syncthreads();
  float bs = 0.0f;
#pragma unroll
  for (int w = 0; w < 8; ++w) bs += sred[w];
  const float inv = 1.0f / bs;

  for (int i = tid; i < Lsz; i += 256) p[i] = __expf(p[i] - bm) * inv;
}

// ---------------------------------------------------------------------------
extern "C" void kernel_launch(void* const* d_in, const int* in_sizes, int n_in,
                              void* d_out, int out_size, void* d_ws, size_t ws_size,
                              hipStream_t stream)
{
  (void)in_sizes; (void)n_in; (void)out_size; (void)ws_size;

  const float* key_in   = (const float*)d_in[0];
  const float* value_in = (const float*)d_in[1];
  const float* query_in = (const float*)d_in[2];
  const unsigned char* mask = (const unsigned char*)d_in[3];
  const float* Wq = (const float*)d_in[4];
  const float* bq = (const float*)d_in[5];
  const float* Wk = (const float*)d_in[6];
  const float* bk = (const float*)d_in[7];
  const float* Wv = (const float*)d_in[8];
  const float* bv = (const float*)d_in[9];
  const float* Wo = (const float*)d_in[10];
  const float* bo = (const float*)d_in[11];

  float* out      = (float*)d_out;
  float* attn_out = out + (size_t)Bsz * Lsz * Dsz;   // top_attn region (B,L,L)

  const size_t bufBytes = (size_t)Mrows * Dsz * sizeof(_Float16);  // 8 MB each
  char* ws = (char*)d_ws;
  _Float16* Qb   = (_Float16*)(ws);
  _Float16* Kb   = (_Float16*)(ws + bufBytes);
  _Float16* Vt   = (_Float16*)(ws + 2 * bufBytes);   // transposed (D x B*L)
  _Float16* ctxb = (_Float16*)(ws + 3 * bufBytes);

  const dim3 gg(Dsz / 128, Mrows / 128);             // (8, 32)
  const float qscale = 0.125f;                       // 1/sqrt(DH)

  gemm_bias_kernel<false, false, false><<<gg, 256, 0, stream>>>(
      query_in, Wq, bq, Qb, Mrows, Dsz, Dsz, qscale);
  gemm_bias_kernel<false, false, false><<<gg, 256, 0, stream>>>(
      key_in, Wk, bk, Kb, Mrows, Dsz, Dsz, 1.0f);
  gemm_bias_kernel<false, false, true><<<gg, 256, 0, stream>>>(
      value_in, Wv, bv, Vt, Mrows, Dsz, Dsz, 1.0f);

  const dim3 gf(Lsz / 64, Hsz, Bsz);                 // (32, 16, 2)
  flash_attn_kernel<<<gf, 128, 0, stream>>>(Qb, Kb, Vt, mask, ctxb, attn_out);

  softmax_rows_kernel<<<Bsz * Lsz, 256, 0, stream>>>(attn_out);

  gemm_bias_kernel<true, true, false><<<gg, 256, 0, stream>>>(
      ctxb, Wo, bo, out, Mrows, Dsz, Dsz, 1.0f);
}
